// GCN_50792283243092
// MI455X (gfx1250) — matrix operand order repaired and checked by reading, not compiled, not run on hardware
//
#include <hip/hip_runtime.h>
#include <hip/hip_bf16.h>

#define GD 128  // feature dimension

typedef __attribute__((ext_vector_type(16))) __bf16 v16bf;
typedef __attribute__((ext_vector_type(8)))  float  v8f;

union V16 { v16bf v; __bf16 e[16]; };
union V8  { v8f   v; float  e[8];  };

// ---------------- utility: zero a float buffer ----------------
__global__ __launch_bounds__(256) void k_zero(float* __restrict__ p, long long n) {
  long long i = (long long)blockIdx.x * blockDim.x + threadIdx.x;
  long long stride = (long long)gridDim.x * blockDim.x;
  for (; i < n; i += stride) p[i] = 0.0f;
}

// ---------------- degree accumulation (f32 atomics, L2-resident) ----------------
__global__ __launch_bounds__(256)
void k_degrees(const int* __restrict__ src, const int* __restrict__ dst,
               float* __restrict__ dout, float* __restrict__ din, int E) {
  int e = blockIdx.x * blockDim.x + threadIdx.x;
  if (e >= E) return;
  __hip_atomic_fetch_add(&dout[src[e]], 1.0f, __ATOMIC_RELAXED, __HIP_MEMORY_SCOPE_AGENT);
  __hip_atomic_fetch_add(&din[dst[e]],  1.0f, __ATOMIC_RELAXED, __HIP_MEMORY_SCOPE_AGENT);
}

// rsqrt(clip(deg,1)) in place (applied to the contiguous [dout|din] region)
__global__ __launch_bounds__(256) void k_rsqrt_clip(float* __restrict__ p, int n) {
  int i = blockIdx.x * blockDim.x + threadIdx.x;
  if (i < n) p[i] = rsqrtf(fmaxf(p[i], 1.0f));
}

// ---------------- GEMM: H = transform(X) @ W via bf16x3-split WMMA ----------------
// transform(X)[r,c] = relu?(X[r,c]*rdi[r] + bias[c]) when rdi != nullptr (fused
// finalize of the previous layer), identity otherwise.
// Block = 256 threads = 8 waves; each block computes a 128x128 output tile.
__global__ __launch_bounds__(256)
void k_gemm_bf16x3(const float* __restrict__ X, const float* __restrict__ W,
                   float* __restrict__ H, int nrows,
                   const float* __restrict__ rdi, const float* __restrict__ bias,
                   int act) {
  // W transposed into LDS as bf16 hi/lo pair: Wt[j*128 + k] = W[k*128 + j]
  __shared__ __align__(32) __bf16 WtH[GD * GD];
  __shared__ __align__(32) __bf16 WtL[GD * GD];

  const int tid = threadIdx.x;
  for (int idx = tid; idx < GD * GD; idx += 256) {
    int k = idx >> 7, j = idx & (GD - 1);
    float w = W[idx];
    __bf16 wh = (__bf16)w;
    WtH[j * GD + k] = wh;
    WtL[j * GD + k] = (__bf16)(w - (float)wh);
  }
  __syncthreads();

  const int wave = tid >> 5;        // 0..7  (wave32)
  const int lane = tid & 31;
  const int l16  = lane & 15;
  const int hig  = lane >> 4;       // 0/1 half-wave group
  const int rowBase = blockIdx.x * 128 + wave * 16;
  const int arow = rowBase + l16;   // A-matrix row for this lane (ISA 16-bit A layout)
  const bool rowOK = arow < nrows;
  const float* xrow = X + (size_t)(rowOK ? arow : 0) * GD;
  const float nd = (rowOK && rdi) ? rdi[arow] : 1.0f;

  v8f acc[8];
  {
    V8 z;
    #pragma unroll
    for (int i = 0; i < 8; ++i) z.e[i] = 0.0f;
    #pragma unroll
    for (int t = 0; t < 8; ++t) acc[t] = z.v;
  }

  #pragma unroll
  for (int kt = 0; kt < 4; ++kt) {            // K in steps of 32
    const int k0 = kt * 32;
    const int c0 = k0 + hig * 16;             // lane's contiguous K-chunk: [c0,c0+16)
    float tmp[16];
    if (rowOK) {
      const float4 a0 = *(const float4*)(xrow + c0);
      const float4 a1 = *(const float4*)(xrow + c0 + 4);
      const float4 a2 = *(const float4*)(xrow + c0 + 8);
      const float4 a3 = *(const float4*)(xrow + c0 + 12);
      tmp[0]=a0.x; tmp[1]=a0.y; tmp[2]=a0.z; tmp[3]=a0.w;
      tmp[4]=a1.x; tmp[5]=a1.y; tmp[6]=a1.z; tmp[7]=a1.w;
      tmp[8]=a2.x; tmp[9]=a2.y; tmp[10]=a2.z; tmp[11]=a2.w;
      tmp[12]=a3.x; tmp[13]=a3.y; tmp[14]=a3.z; tmp[15]=a3.w;
      if (rdi) {  // fused: previous layer's norm_dst scale + bias + ReLU
        #pragma unroll
        for (int i = 0; i < 16; ++i) {
          int c = c0 + i;
          float v = tmp[i] * nd + bias[c];
          tmp[i] = act ? fmaxf(v, 0.0f) : v;
        }
      }
    } else {
      #pragma unroll
      for (int i = 0; i < 16; ++i) tmp[i] = 0.0f;
    }

    V16 ah, al;   // bf16 hi/lo split of the A fragment
    #pragma unroll
    for (int i = 0; i < 16; ++i) {
      __bf16 h = (__bf16)tmp[i];
      ah.e[i] = h;
      al.e[i] = (__bf16)(tmp[i] - (float)h);
    }

    #pragma unroll
    for (int nt = 0; nt < 8; ++nt) {          // 8 N-tiles of 16 columns
      // B fragment (32x16): lane -> column nt*16+l16; 16 contiguous K bf16 starting
      // at k0 + hig*16 (contiguous k = 16*hig + s, same order as A; column-major per lane)
      const int boff = (nt * 16 + l16) * GD + k0 + hig * 16;
      v16bf bh = *(const v16bf*)(WtH + boff);
      v16bf bl = *(const v16bf*)(WtL + boff);
      acc[nt] = __builtin_amdgcn_wmma_f32_16x16x32_bf16(
          false, ah.v, false, bh, (short)0, acc[nt], false, false);
      acc[nt] = __builtin_amdgcn_wmma_f32_16x16x32_bf16(
          false, al.v, false, bh, (short)0, acc[nt], false, false);
      acc[nt] = __builtin_amdgcn_wmma_f32_16x16x32_bf16(
          false, ah.v, false, bl, (short)0, acc[nt], false, false);
    }
  }

  // D layout: element (M,N) -> lane = N%16 + 16*(M%16>=8), vgpr r = M%8
  #pragma unroll
  for (int nt = 0; nt < 8; ++nt) {
    V8 a; a.v = acc[nt];
    #pragma unroll
    for (int r = 0; r < 8; ++r) {
      int row = rowBase + hig * 8 + r;
      if (row < nrows) H[(size_t)row * GD + nt * 16 + l16] = a.e[r];
    }
  }
}

// ---------------- edge scatter: AGG[dst] += H[src] * rsqrt(deg_out[src]) ----------------
// One wave per edge; 32 lanes x float4 = 128 features. H/AGG are L2-resident (51 MB each).
__global__ __launch_bounds__(256)
void k_scatter(const float* __restrict__ H, const int* __restrict__ src,
               const int* __restrict__ dst, const float* __restrict__ rdo,
               float* __restrict__ AGG, int E) {
  int e = blockIdx.x * 8 + (threadIdx.x >> 5);
  if (e >= E) return;
  int lane = threadIdx.x & 31;
  int s = src[e], d = dst[e];
  float ns = rdo[s];
  float4 v = *((const float4*)(H + (size_t)s * GD) + lane);
  float* ap = AGG + (size_t)d * GD + lane * 4;
  __hip_atomic_fetch_add(ap + 0, v.x * ns, __ATOMIC_RELAXED, __HIP_MEMORY_SCOPE_AGENT);
  __hip_atomic_fetch_add(ap + 1, v.y * ns, __ATOMIC_RELAXED, __HIP_MEMORY_SCOPE_AGENT);
  __hip_atomic_fetch_add(ap + 2, v.z * ns, __ATOMIC_RELAXED, __HIP_MEMORY_SCOPE_AGENT);
  __hip_atomic_fetch_add(ap + 3, v.w * ns, __ATOMIC_RELAXED, __HIP_MEMORY_SCOPE_AGENT);
}

// ---------------- final layer epilogue: out = AGG*rdi + b (no activation) ----------------
__global__ __launch_bounds__(256)
void k_finalize(const float* __restrict__ AGG, const float* __restrict__ rdi,
                const float* __restrict__ b, float* __restrict__ out,
                int nrows, int act) {
  int i = blockIdx.x * blockDim.x + threadIdx.x;   // over nrows*32 float4s
  if (i >= nrows * 32) return;
  int node = i >> 5;
  int q = i & 31;
  float nd = rdi[node];
  float4 v  = ((const float4*)AGG)[i];
  float4 bb = ((const float4*)b)[q];
  float4 o;
  o.x = v.x * nd + bb.x;
  o.y = v.y * nd + bb.y;
  o.z = v.z * nd + bb.z;
  o.w = v.w * nd + bb.w;
  if (act) {
    o.x = fmaxf(o.x, 0.0f); o.y = fmaxf(o.y, 0.0f);
    o.z = fmaxf(o.z, 0.0f); o.w = fmaxf(o.w, 0.0f);
  }
  ((float4*)out)[i] = o;
}

extern "C" void kernel_launch(void* const* d_in, const int* in_sizes, int n_in,
                              void* d_out, int out_size, void* d_ws, size_t ws_size,
                              hipStream_t stream) {
  const float* feats = (const float*)d_in[0];
  const float* W1 = (const float*)d_in[1];
  const float* b1 = (const float*)d_in[2];
  const float* W2 = (const float*)d_in[3];
  const float* b2 = (const float*)d_in[4];
  const float* W3 = (const float*)d_in[5];
  const float* b3 = (const float*)d_in[6];
  const int*   src = (const int*)d_in[7];
  const int*   dst = (const int*)d_in[8];

  const int N = in_sizes[0] / GD;   // 100000
  const int E = in_sizes[7];        // 1,700,000

  // Workspace layout (floats): H | AGGa | AGGb | rdeg_out | rdeg_in
  float* H    = (float*)d_ws;
  float* AGGa = H    + (size_t)N * GD;
  float* AGGb = AGGa + (size_t)N * GD;
  float* rdo  = AGGb + (size_t)N * GD;
  float* rdi  = rdo + N;

  const long long ND = (long long)N * GD;
  const int gemmBlocks = (N + 127) / 128;
  const int scatBlocks = (E + 7) / 8;
  const int finBlocks  = (N * 32 + 255) / 256;

  // degrees -> rsqrt(clip(deg,1)); rdo/rdi are contiguous 2N floats
  k_zero<<<256, 256, 0, stream>>>(rdo, 2LL * N);
  k_degrees<<<(E + 255) / 256, 256, 0, stream>>>(src, dst, rdo, rdi, E);
  k_rsqrt_clip<<<(2 * N + 255) / 256, 256, 0, stream>>>(rdo, 2 * N);

  // Layer 1: H = feats @ W1 ; AGGa = scatter(H)
  k_gemm_bf16x3<<<gemmBlocks, 256, 0, stream>>>(feats, W1, H, N, nullptr, nullptr, 0);
  k_zero<<<2048, 256, 0, stream>>>(AGGa, ND);
  k_scatter<<<scatBlocks, 256, 0, stream>>>(H, src, dst, rdo, AGGa, E);

  // Layer 2: H = relu(AGGa*rdi + b1) @ W2 (epilogue fused into A load); AGGb = scatter(H)
  k_gemm_bf16x3<<<gemmBlocks, 256, 0, stream>>>(AGGa, W2, H, N, rdi, b1, 1);
  k_zero<<<2048, 256, 0, stream>>>(AGGb, ND);
  k_scatter<<<scatBlocks, 256, 0, stream>>>(H, src, dst, rdo, AGGb, E);

  // Layer 3: H = relu(AGGb*rdi + b2) @ W3 ; AGGa = scatter(H)
  k_gemm_bf16x3<<<gemmBlocks, 256, 0, stream>>>(AGGb, W3, H, N, rdi, b2, 1);
  k_zero<<<2048, 256, 0, stream>>>(AGGa, ND);
  k_scatter<<<scatBlocks, 256, 0, stream>>>(H, src, dst, rdo, AGGa, E);

  // Final epilogue (no activation) straight to d_out
  k_finalize<<<finBlocks, 256, 0, stream>>>(AGGa, rdi, b3, (float*)d_out, N, 0);
}
